// HierarchicalAttention_38611755991183
// MI455X (gfx1250) — matrix operand order repaired and checked by reading, not compiled, not run on hardware
//
#include <hip/hip_runtime.h>

typedef __attribute__((ext_vector_type(16))) _Float16 v16h;
typedef __attribute__((ext_vector_type(8)))  _Float16 v8h;
typedef __attribute__((ext_vector_type(8)))  float    v8f;

#define TOK_    32768   // B*T = 2*16384
#define D_      512
#define HEADS_  8
#define HD_     64
#define CH_     512
#define NCHUNK_ 64      // TOK_/CH_

static __device__ inline v8f vzero8() {
  v8f z;
#pragma unroll
  for (int i = 0; i < 8; ++i) z[i] = 0.0f;
  return z;
}

// A-matrix 16x32 f16 fragment (M rows, K contiguous, stride ld elements).
// Per ISA: lanes 0-15 hold K[0..7] (v0-3) and K[16..23] (v4-7); lanes 16-31
// hold K[8..15] and K[24..31].
static __device__ inline v16h load_fragA(const _Float16* __restrict__ base, int ld) {
  const int lane = threadIdx.x & 31;
  const _Float16* p = base + (size_t)(lane & 15) * ld + ((lane >> 4) << 3);
  union { v16h v; v8h h[2]; } u;
  u.h[0] = *(const v8h*)p;
  u.h[1] = *(const v8h*)(p + 16);
  return u.v;
}

// B-matrix 32x16 f16 fragment, sourced from B^T stored N x K row-major
// (K contiguous). Same K order as A: lane = N, lanes 0-15 hold K[0..7] and
// K[16..23]; lanes 16-31 hold K[8..15] and K[24..31].
static __device__ inline v16h load_fragB(const _Float16* __restrict__ base, int ld) {
  const int lane = threadIdx.x & 31;
  const _Float16* p = base + (size_t)(lane & 15) * ld + ((lane >> 4) << 3);
  union { v16h v; v8h h[2]; } u;
  u.h[0] = *(const v8h*)p;
  u.h[1] = *(const v8h*)(p + 16);
  return u.v;
}

static __device__ inline v8f wmma16(v16h a, v16h b, v8f c) {
  return __builtin_amdgcn_wmma_f32_16x16x32_f16(false, a, false, b, (short)0, c,
                                                false, false);
}

// ---------------------------------------------------------------- converts
__global__ void cvt_f32_f16(const float* __restrict__ in, _Float16* __restrict__ out, int n) {
  int i = blockIdx.x * blockDim.x + threadIdx.x;
  if (i < n) out[i] = (_Float16)in[i];
}

// ------------------------------------------------------------- WMMA GEMM
// C(MxN) = A(MxK,f16) @ Bt(NxK,f16)^T, f32 accumulate.
// MODE 0: store f16 row-major.  MODE 1: same, scaled by 0.125 (Q).
// MODE 2: store f16 transposed per 512-token chunk: C[(chunk*512+n)*512 + m%512]
template <int MODE>
__global__ __launch_bounds__(128) void gemm_xw(const _Float16* __restrict__ A,
                                               const _Float16* __restrict__ Bt,
                                               _Float16* __restrict__ C,
                                               int M, int N, int K) {
  const int wave = threadIdx.x >> 5;
  const int m0 = blockIdx.x * 128 + (wave & 1) * 64;
  const int n0 = blockIdx.y * 128 + (wave >> 1) * 64;

  v8f acc[4][4];
  for (int i = 0; i < 4; ++i)
    for (int j = 0; j < 4; ++j) acc[i][j] = vzero8();

  for (int k0 = 0; k0 < K; k0 += 32) {
    if (k0 + 32 < K) {
      __builtin_prefetch(A + (size_t)m0 * K + k0 + 32, 0, 1);
      __builtin_prefetch(Bt + (size_t)n0 * K + k0 + 32, 0, 1);
    }
    v16h a[4], b[4];
#pragma unroll
    for (int i = 0; i < 4; ++i)
      a[i] = load_fragA(A + (size_t)(m0 + 16 * i) * K + k0, K);
#pragma unroll
    for (int j = 0; j < 4; ++j)
      b[j] = load_fragB(Bt + (size_t)(n0 + 16 * j) * K + k0, K);
#pragma unroll
    for (int i = 0; i < 4; ++i)
#pragma unroll
      for (int j = 0; j < 4; ++j)
        acc[i][j] = wmma16(a[i], b[j], acc[i][j]);
  }

  const int lane = threadIdx.x & 31;
  const int n_lane = lane & 15;
  const int mb = (lane >> 4) << 3;  // C/D frag: vgpr r -> row mb+r, lane%16 -> col
#pragma unroll
  for (int i = 0; i < 4; ++i) {
#pragma unroll
    for (int j = 0; j < 4; ++j) {
      if (MODE == 2) {
        const int mrow = m0 + 16 * i + mb;        // 8 consecutive tokens
        const int chunk = mrow >> 9;
        const int tl = mrow & 511;
        const int feat = n0 + 16 * j + n_lane;
        union { v8h v; _Float16 e[8]; } u;
#pragma unroll
        for (int r = 0; r < 8; ++r) u.e[r] = (_Float16)acc[i][j][r];
        *(v8h*)(C + ((size_t)chunk * 512 + feat) * 512 + tl) = u.v;
      } else {
#pragma unroll
        for (int r = 0; r < 8; ++r) {
          float v = acc[i][j][r];
          if (MODE == 1) v *= 0.125f;  // 1/sqrt(hd)
          const int m = m0 + 16 * i + mb + r;
          const int n = n0 + 16 * j + n_lane;
          C[(size_t)m * N + n] = (_Float16)v;
        }
      }
    }
  }
}

// --------------------------------------------------- per-chunk attention
// One wave per (16-row q-tile, head, chunk). S=Q K^T (WMMA) -> LDS f32,
// two-pass softmax stats, O = P V via WMMA with P built on the fly.
__global__ __launch_bounds__(32) void attn_chunk(const _Float16* __restrict__ Qh,
                                                 const _Float16* __restrict__ Kh,
                                                 const _Float16* __restrict__ Vt,
                                                 _Float16* __restrict__ AO) {
  const int qt = blockIdx.x;    // 0..31 (16-row tiles)
  const int head = blockIdx.y;  // 0..7
  const int chunk = blockIdx.z; // 0..63
  const int lane = threadIdx.x;

  __shared__ float S[16][516];
  __shared__ float pmax[2][16];
  __shared__ float psum[2][16];
  __shared__ float rmax[16];
  __shared__ float rinv[16];

  const size_t tokBase = (size_t)chunk * 512 + (size_t)qt * 16;
  const _Float16* Qbase = Qh + tokBase * 512 + head * HD_;
  const v16h qf0 = load_fragA(Qbase, 512);       // hd 0..31  (scale pre-folded)
  const v16h qf1 = load_fragA(Qbase + 32, 512);  // hd 32..63

  const int n_lane = lane & 15;
  const int mb = (lane >> 4) << 3;

  for (int kb = 0; kb < 8; ++kb) {  // key blocks of 64
    v8f s[4];
    for (int nj = 0; nj < 4; ++nj) s[nj] = vzero8();
    const _Float16* Kb = Kh + ((size_t)chunk * 512 + kb * 64) * 512 + head * HD_;
#pragma unroll
    for (int nj = 0; nj < 4; ++nj) {
      v16h b0 = load_fragB(Kb + (size_t)nj * 16 * 512, 512);
      s[nj] = wmma16(qf0, b0, s[nj]);
      v16h b1 = load_fragB(Kb + (size_t)nj * 16 * 512 + 32, 512);
      s[nj] = wmma16(qf1, b1, s[nj]);
    }
#pragma unroll
    for (int nj = 0; nj < 4; ++nj)
#pragma unroll
      for (int r = 0; r < 8; ++r)
        S[mb + r][kb * 64 + nj * 16 + n_lane] = s[nj][r];
  }
  __syncthreads();

  {  // row max / sum-of-exp, each lane owns half a row
    const int row = lane & 15;
    const int half = lane >> 4;
    const float* sr = &S[row][half * 256];
    float mx = -1e30f;
    for (int c = 0; c < 256; ++c) mx = fmaxf(mx, sr[c]);
    pmax[half][row] = mx;
    __syncthreads();
    const float gmx = fmaxf(pmax[0][row], pmax[1][row]);
    float sum = 0.f;
    for (int c = 0; c < 256; ++c) sum += __expf(sr[c] - gmx);
    psum[half][row] = sum;
    __syncthreads();
    if (half == 0) {
      rmax[row] = gmx;
      rinv[row] = 1.0f / (psum[0][row] + psum[1][row]);
    }
  }
  __syncthreads();

  v8f o[4];
  for (int nj = 0; nj < 4; ++nj) o[nj] = vzero8();
  const int m_a = lane & 15;
  const int kh = lane >> 4;
  const float mxv = rmax[m_a];
  for (int jb = 0; jb < 512; jb += 32) {  // keys
    union { v16h v; _Float16 e[16]; } pu;  // P in A-fragment layout
    const float* s0 = &S[m_a][jb + kh * 8];
#pragma unroll
    for (int e = 0; e < 8; ++e) pu.e[e] = (_Float16)__expf(s0[e] - mxv);
#pragma unroll
    for (int e = 0; e < 8; ++e) pu.e[8 + e] = (_Float16)__expf(s0[16 + e] - mxv);
    const _Float16* Vb = Vt + ((size_t)chunk * 512 + head * HD_) * 512 + jb;
#pragma unroll
    for (int nj = 0; nj < 4; ++nj) {
      v16h vf = load_fragB(Vb + (size_t)nj * 16 * 512, 512);
      o[nj] = wmma16(pu.v, vf, o[nj]);
    }
  }
#pragma unroll
  for (int nj = 0; nj < 4; ++nj)
#pragma unroll
    for (int r = 0; r < 8; ++r) {
      const int row = mb + r;
      AO[(tokBase + row) * 512 + head * HD_ + nj * 16 + n_lane] =
          (_Float16)(o[nj][r] * rinv[row]);
    }
}

// ------------------------------------------------------- chunk summaries
__global__ __launch_bounds__(512) void summarize(const _Float16* __restrict__ LO,
                                                 float* __restrict__ SUM) {
  const int c = blockIdx.x;       // chunk 0..63
  const int f = threadIdx.x;      // feature
  const _Float16* p = LO + (size_t)c * CH_ * D_ + f;
  float s = 0.f;
  for (int t = 0; t < CH_; ++t) s += (float)p[(size_t)t * D_];
  SUM[c * D_ + f] = s * (1.0f / CH_);
}

// -------------------------------------------- cross-chunk (global) stage
__global__ __launch_bounds__(64) void global_attn(const float* __restrict__ SUM,
                                                  const float* __restrict__ Wg,
                                                  float* __restrict__ G) {
  const int b = blockIdx.x >> 3;
  const int h = blockIdx.x & 7;
  __shared__ float g[32][HD_];
  __shared__ float at[32][33];
  const int tid = threadIdx.x;

  for (int idx = tid; idx < 32 * HD_; idx += 64) {  // g = summaries @ Wg^T (head slice)
    const int j = idx >> 6, dd = idx & 63;
    const float* sr = SUM + ((size_t)b * 32 + j) * D_;
    const float* wr = Wg + (size_t)(h * HD_ + dd) * D_;
    float s = 0.f;
    for (int k = 0; k < D_; ++k) s += sr[k] * wr[k];
    g[j][dd] = s;
  }
  __syncthreads();

  if (tid < 32) {  // 32x32 softmax attention, one row per lane
    float row[32];
    float mx = -1e30f;
    for (int j = 0; j < 32; ++j) {
      float s = 0.f;
      for (int d = 0; d < HD_; ++d) s += g[tid][d] * g[j][d];
      s *= 0.125f;
      row[j] = s;
      mx = fmaxf(mx, s);
    }
    float sum = 0.f;
    for (int j = 0; j < 32; ++j) { row[j] = __expf(row[j] - mx); sum += row[j]; }
    const float inv = 1.0f / sum;
    for (int j = 0; j < 32; ++j) at[tid][j] = row[j] * inv;
  }
  __syncthreads();

  for (int idx = tid; idx < 32 * HD_; idx += 64) {
    const int i = idx >> 6, dd = idx & 63;
    float s = 0.f;
    for (int j = 0; j < 32; ++j) s += at[i][j] * g[j][dd];
    G[((size_t)b * 32 + i) * D_ + h * HD_ + dd] = s;
  }
}

// ------------------------------------------------------------ final add
__global__ void final_add(const _Float16* __restrict__ LO, const float* __restrict__ G,
                          float* __restrict__ out, int n) {
  int i = blockIdx.x * blockDim.x + threadIdx.x;
  if (i >= n) return;
  const int f = i & 511;
  const int chunk = i >> 18;  // (i/512)/512
  out[i] = (float)LO[i] + G[(chunk << 9) + f];
}

extern "C" void kernel_launch(void* const* d_in, const int* in_sizes, int n_in,
                              void* d_out, int out_size, void* d_ws, size_t ws_size,
                              hipStream_t stream) {
  (void)in_sizes; (void)n_in; (void)out_size; (void)ws_size;
  const float* x  = (const float*)d_in[0];
  const float* Wq = (const float*)d_in[1];
  const float* Wk = (const float*)d_in[2];
  const float* Wv = (const float*)d_in[3];
  const float* Wo = (const float*)d_in[4];
  const float* Wg = (const float*)d_in[5];
  float* out = (float*)d_out;

  char* ws = (char*)d_ws;
  size_t off = 0;
  auto carve = [&](size_t bytes) -> char* {
    char* p = ws + off;
    off += (bytes + 255) & ~(size_t)255;
    return p;
  };
  _Float16* xh  = (_Float16*)carve((size_t)TOK_ * D_ * sizeof(_Float16));
  _Float16* Wqh = (_Float16*)carve((size_t)D_ * D_ * sizeof(_Float16));
  _Float16* Wkh = (_Float16*)carve((size_t)D_ * D_ * sizeof(_Float16));
  _Float16* Wvh = (_Float16*)carve((size_t)D_ * D_ * sizeof(_Float16));
  _Float16* Woh = (_Float16*)carve((size_t)D_ * D_ * sizeof(_Float16));
  _Float16* Qh  = (_Float16*)carve((size_t)TOK_ * D_ * sizeof(_Float16));
  _Float16* Kh  = (_Float16*)carve((size_t)TOK_ * D_ * sizeof(_Float16));
  _Float16* Vt  = (_Float16*)carve((size_t)TOK_ * D_ * sizeof(_Float16));
  float* summ = (float*)carve((size_t)NCHUNK_ * D_ * sizeof(float));
  float* gctx = (float*)carve((size_t)NCHUNK_ * D_ * sizeof(float));
  _Float16* AO = xh;  // alias: x(f16) dead after the three projections
  _Float16* LO = Qh;  // alias: Q dead after attention

  const int nX = TOK_ * D_;
  const int nW = D_ * D_;
  cvt_f32_f16<<<(nX + 255) / 256, 256, 0, stream>>>(x, xh, nX);
  cvt_f32_f16<<<(nW + 255) / 256, 256, 0, stream>>>(Wq, Wqh, nW);
  cvt_f32_f16<<<(nW + 255) / 256, 256, 0, stream>>>(Wk, Wkh, nW);
  cvt_f32_f16<<<(nW + 255) / 256, 256, 0, stream>>>(Wv, Wvh, nW);
  cvt_f32_f16<<<(nW + 255) / 256, 256, 0, stream>>>(Wo, Woh, nW);

  dim3 gg(TOK_ / 128, D_ / 128, 1), gb(128);
  gemm_xw<1><<<gg, gb, 0, stream>>>(xh, Wqh, Qh, TOK_, D_, D_);  // Q (pre-scaled)
  gemm_xw<0><<<gg, gb, 0, stream>>>(xh, Wkh, Kh, TOK_, D_, D_);  // K
  gemm_xw<2><<<gg, gb, 0, stream>>>(xh, Wvh, Vt, TOK_, D_, D_);  // V transposed

  attn_chunk<<<dim3(32, HEADS_, NCHUNK_), 32, 0, stream>>>(Qh, Kh, Vt, AO);

  gemm_xw<0><<<gg, gb, 0, stream>>>(AO, Woh, LO, TOK_, D_, D_);  // local_out

  summarize<<<NCHUNK_, 512, 0, stream>>>(LO, summ);
  global_attn<<<16, 64, 0, stream>>>(summ, Wg, gctx);
  final_add<<<(nX + 255) / 256, 256, 0, stream>>>(LO, gctx, out, nX);
}
